// Transformer_72645076844587
// MI455X (gfx1250) — matrix operand order repaired and checked by reading, not compiled, not run on hardware
//
#include <hip/hip_runtime.h>
#include <math.h>

// ---------------- problem constants ----------------
#define C_DIM   768
#define OUT_DIM 768
#define NH      16
#define HD      48
#define B_SZ    32
#define N_TOK   3136
#define BN      (B_SZ * N_TOK)     // 100352 rows
#define GRID_SZ 56                 // sqrt(3136)
#define KSTEPS  24                 // 768 / 32
#define EPS_LN  1e-5f

#define KV_RT   2                  // row-tiles per block in kv kernel (32 rows)
#define OP_RT   4                  // row-tiles per block in o-proj kernel (64 rows)
#define LDS_TILE_B 1280            // 16 rows * 80B (64B data + 16B pad)

typedef __attribute__((ext_vector_type(16))) __bf16 v16bf;
typedef __attribute__((ext_vector_type(8)))  __bf16 v8bf;
typedef __attribute__((ext_vector_type(8)))  float  v8f;

union V16U { v16bf v; v8bf h[2]; };

#define WMMA_BF16(A, Bf, Cc) __builtin_amdgcn_wmma_f32_16x16x32_bf16( \
    false, (A), false, (Bf), (short)0, (Cc), false, false)

// ---- CDNA5 async global->LDS staging (guarded: falls back to direct loads) --
#ifndef __has_builtin
#define __has_builtin(x) 0
#endif
#if __has_builtin(__builtin_amdgcn_global_load_async_to_lds_b128)
#define HAVE_ASYNC_LDS 1
typedef __attribute__((__vector_size__(4 * sizeof(int)))) int v4i_async;
typedef __attribute__((address_space(1))) v4i_async* gv4i_p;
typedef __attribute__((address_space(3))) v4i_async* lv4i_p;
#define ASYNC_CP_B128(g, l) __builtin_amdgcn_global_load_async_to_lds_b128( \
    (gv4i_p)(g), (lv4i_p)(l), 0, 0)
#if __has_builtin(__builtin_amdgcn_s_wait_asynccnt)
#define WAIT_ASYNC0() __builtin_amdgcn_s_wait_asynccnt(0)
#else
#define WAIT_ASYNC0() asm volatile("s_wait_asynccnt 0x0" ::: "memory")
#endif
#else
#define HAVE_ASYNC_LDS 0
#endif

// ---------------------------------------------------------------------------
// K0: fp32 weight (row-major [k][col], 768x768) -> bf16 WMMA-B-fragment-major:
//   frag element addr ((ct*KSTEPS+ks)*32 + lane)*16 + e
//   <- W[ks*32 + (lane>>4)*8 + (e&7) + (e>>3)*16][ct*16 + (lane&15)]
// ---------------------------------------------------------------------------
__global__ void pack_weights(const float* __restrict__ w, __bf16* __restrict__ wf) {
    int idx  = blockIdx.x * 256 + threadIdx.x;      // 0 .. 589823
    int e    = idx & 15;
    int lane = (idx >> 4) & 31;
    int rest = idx >> 9;                            // ct*KSTEPS + ks
    int ks   = rest % KSTEPS;
    int ct   = rest / KSTEPS;
    int kk   = ks * 32 + (lane >> 4) * 8 + (e & 7) + (e >> 3) * 16;
    int col  = ct * 16 + (lane & 15);
    wf[idx] = (__bf16)w[(size_t)kk * OUT_DIM + col];
}

// ---------------------------------------------------------------------------
// block-wide mean / rstd over 768 elements (3 per thread, 256 threads)
// ---------------------------------------------------------------------------
__device__ inline void blk_meanvar(float v0, float v1, float v2, float* red,
                                   int tid, float& mean, float& rstd) {
    float s  = v0 + v1 + v2;
    float s2 = v0 * v0 + v1 * v1 + v2 * v2;
    #pragma unroll
    for (int m = 1; m < 32; m <<= 1) {
        s  += __shfl_xor(s,  m, 32);
        s2 += __shfl_xor(s2, m, 32);
    }
    int wv = tid >> 5;
    if ((tid & 31) == 0) { red[wv] = s; red[wv + 8] = s2; }
    __syncthreads();
    if (tid == 0) {
        float ts = 0.f, t2 = 0.f;
        for (int i = 0; i < 8; i++) { ts += red[i]; t2 += red[i + 8]; }
        float mn  = ts * (1.f / 768.f);
        float var = t2 * (1.f / 768.f) - mn * mn;
        red[0] = mn;
        red[1] = rsqrtf(var + EPS_LN);
    }
    __syncthreads();
    mean = red[0];
    rstd = red[1];
}

// ---------------------------------------------------------------------------
// K1: layernorm(embeddings) -> bf16 row-major (one block per row)
// ---------------------------------------------------------------------------
__global__ void __launch_bounds__(256)
ln_to_bf16(const float* __restrict__ x, const float* __restrict__ g,
           const float* __restrict__ be, __bf16* __restrict__ y) {
    __shared__ float red[16];
    int row = blockIdx.x;
    int tid = threadIdx.x;
    const float* xr = x + (size_t)row * C_DIM;
    float v0 = xr[tid], v1 = xr[tid + 256], v2 = xr[tid + 512];
    float mean, rstd;
    blk_meanvar(v0, v1, v2, red, tid, mean, rstd);
    __bf16* yr = y + (size_t)row * C_DIM;
    yr[tid]       = (__bf16)((v0 - mean) * rstd * g[tid]       + be[tid]);
    yr[tid + 256] = (__bf16)((v1 - mean) * rstd * g[tid + 256] + be[tid + 256]);
    yr[tid + 512] = (__bf16)((v2 - mean) * rstd * g[tid + 512] + be[tid + 512]);
}

// ---------------------------------------------------------------------------
// K2: per-batch reference token -> LN -> MLP -> ref, q  (fp32 exact, tiny)
// ---------------------------------------------------------------------------
__global__ void __launch_bounds__(256)
ref_mlp_q(const float* __restrict__ emb, const int* __restrict__ rloc,
          const float* __restrict__ lnw, const float* __restrict__ lnb,
          const float* __restrict__ w1, const float* __restrict__ b1,
          const float* __restrict__ w2, const float* __restrict__ b2,
          const float* __restrict__ wq, const float* __restrict__ bq,
          float* __restrict__ out_ref, float* __restrict__ out_q,
          float* __restrict__ q_ws) {
    __shared__ float sref[772];
    __shared__ float sh1[1536];
    __shared__ float red[16];
    const int b = blockIdx.x, tid = threadIdx.x;
    const int rx = rloc[b * 2 + 0], ry = rloc[b * 2 + 1];
    const int px = rx / 8, py = ry / 8;
    const int flat = px * GRID_SZ + py;
    const float* xr = emb + ((size_t)b * N_TOK + flat) * C_DIM;
    float v0 = xr[tid], v1 = xr[tid + 256], v2 = xr[tid + 512];
    float mean, rstd;
    blk_meanvar(v0, v1, v2, red, tid, mean, rstd);
    sref[tid]       = (v0 - mean) * rstd * lnw[tid]       + lnb[tid];
    sref[tid + 256] = (v1 - mean) * rstd * lnw[tid + 256] + lnb[tid + 256];
    sref[tid + 512] = (v2 - mean) * rstd * lnw[tid + 512] + lnb[tid + 512];
    if (tid == 0) {
        float cx = (float)px * 8.f + 3.5f;
        float cy = (float)py * 8.f + 3.5f;
        sref[768] = ((float)rx - cx) / 3.5001f;   // (s-1)/2 + 1e-4
        sref[769] = ((float)ry - cy) / 3.5001f;
    }
    __syncthreads();
    for (int j = tid; j < 1536; j += 256) {           // h1 = relu(ref @ w1 + b1)
        float acc = b1[j];
        for (int i = 0; i < 770; i++) acc += sref[i] * w1[(size_t)i * 1536 + j];
        sh1[j] = fmaxf(acc, 0.f);
    }
    __syncthreads();
    for (int c = tid; c < 768; c += 256) {            // ref = h1 @ w2 + b2
        float acc = b2[c];
        for (int j = 0; j < 1536; j++) acc += sh1[j] * w2[(size_t)j * 768 + c];
        out_ref[(size_t)b * 768 + c] = acc;
        sref[c] = acc;                                // reads sh1 only -> safe
    }
    __syncthreads();
    for (int c = tid; c < 768; c += 256) {            // q = ref @ wq + bq
        float acc = bq[c];
        for (int i = 0; i < 768; i++) acc += sref[i] * wq[(size_t)i * 768 + c];
        out_q[(size_t)b * 768 + c] = acc;
        q_ws[(size_t)b * 768 + c]  = acc;
    }
}

// ---------------------------------------------------------------------------
// A-fragment loaders (ISA 7.12.2, 16-bit A 16x32):
//   lane (half=lane>>4, m=lane&15): row = m,
//   half 0: K = {0..7,16..23}; half 1: K = {8..15,24..31}
// ---------------------------------------------------------------------------
__device__ inline v16bf load_a_frag_g(const __bf16* base_row, int k0, int half) {
    const __bf16* p = base_row + k0 + half * 8;       // two contiguous 16B loads
    V16U u;
    u.h[0] = *(const v8bf*)(p);
    u.h[1] = *(const v8bf*)(p + 16);
    return u.v;
}
__device__ inline v16bf load_a_frag_lds(const char* tile, int n, int half) {
    const char* p = tile + n * 80 + half * 16;        // 80B padded rows: no bank conflicts
    V16U u;
    u.h[0] = *(const v8bf*)(p);
    u.h[1] = *(const v8bf*)(p + 32);
    return u.v;
}

// ---------------------------------------------------------------------------
// K3: fused k/v projection + sigmoid attention + xpre = attn*v
//   512 threads = 16 waves; wave w == head w (48 cols), KV_RT row-tiles.
//   A tiles staged to LDS via async copies (double buffered), B fragments
//   stream from L2 and are reused across KV_RT row-tiles.
// ---------------------------------------------------------------------------
__global__ void __launch_bounds__(512)
kv_attn(__bf16* ab,                               // emb_bf in, xpre out (aliased)
        const __bf16* __restrict__ wk_f, const __bf16* __restrict__ wv_f,
        const float* __restrict__ bk, const float* __restrict__ bv,
        const float* __restrict__ q_ws,
        float* __restrict__ out_k, float* __restrict__ out_v,
        float* __restrict__ out_attn) {
    const int rb   = blockIdx.x;
    const int row0 = rb * (KV_RT * 16);
    const int b    = row0 / N_TOK;
    const int n0   = row0 % N_TOK;                 // 3136 % 32 == 0: no b crossing
    const int tid  = threadIdx.x;
    const int wave = tid >> 5;                     // == head
    const int lane = tid & 31;
    const int half = lane >> 4;
    const int n    = lane & 15;

    v8f accK[KV_RT][3] = {}, accV[KV_RT][3] = {};

#if HAVE_ASYNC_LDS
    __shared__ __align__(16) char abuf[2][KV_RT * LDS_TILE_B];
    int groff[KV_RT * 2], ldsoff[KV_RT * 2];
    #pragma unroll
    for (int i = 0; i < KV_RT * 2; i++) {          // 16B chunk map (wave 0 stages)
        int c = i * 32 + lane;
        int rt = c >> 6, w = c & 63, r = w >> 2, cc = w & 3;
        groff[i]  = (rt * 16 + r) * (C_DIM * 2) + cc * 16;
        ldsoff[i] = rt * LDS_TILE_B + r * 80 + cc * 16;
    }
    const char* gbase = (const char*)(ab + (size_t)row0 * C_DIM);
    if (wave == 0) {
        #pragma unroll
        for (int i = 0; i < KV_RT * 2; i++)
            ASYNC_CP_B128(gbase + groff[i], &abuf[0][ldsoff[i]]);
    }
#endif

    for (int ks = 0; ks < KSTEPS; ks++) {
#if HAVE_ASYNC_LDS
        const int cur = ks & 1;
        if (wave == 0) WAIT_ASYNC0();
        __syncthreads();                            // buf[cur] ready, buf[cur^1] free
        if (wave == 0 && ks + 1 < KSTEPS) {
            #pragma unroll
            for (int i = 0; i < KV_RT * 2; i++)
                ASYNC_CP_B128(gbase + groff[i] + (ks + 1) * 64,
                              &abuf[cur ^ 1][ldsoff[i]]);
        }
#endif
        v16bf bkf[3], bvf[3];
        #pragma unroll
        for (int t = 0; t < 3; t++) {
            size_t fo = ((size_t)((wave * 3 + t) * KSTEPS + ks) * 32 + lane) * 16;
            bkf[t] = *(const v16bf*)(wk_f + fo);
            bvf[t] = *(const v16bf*)(wv_f + fo);
        }
        #pragma unroll
        for (int rt = 0; rt < KV_RT; rt++) {
#if HAVE_ASYNC_LDS
            v16bf af = load_a_frag_lds(&abuf[cur][rt * LDS_TILE_B], n, half);
#else
            v16bf af = load_a_frag_g(ab + (size_t)(row0 + rt * 16 + n) * C_DIM,
                                     ks * 32, half);
#endif
            #pragma unroll
            for (int t = 0; t < 3; t++) {
                accK[rt][t] = WMMA_BF16(af, bkf[t], accK[rt][t]);
                accV[rt][t] = WMMA_BF16(af, bvf[t], accV[rt][t]);
            }
        }
    }

    // -------- epilogue: bias, attn = sigmoid(scale * q.k), xpre = attn*v -----
    float qv[3];
    #pragma unroll
    for (int t = 0; t < 3; t++) {
        int col = wave * 48 + t * 16 + n;
        float bkc = bk[col], bvc = bv[col];
        qv[t] = q_ws[(size_t)b * 768 + col];
        #pragma unroll
        for (int rt = 0; rt < KV_RT; rt++)
            #pragma unroll
            for (int j = 0; j < 8; j++) { accK[rt][t][j] += bkc; accV[rt][t][j] += bvc; }
    }
    const float scl = 0.14433756729740643f;        // 48^-0.5
    #pragma unroll
    for (int rt = 0; rt < KV_RT; rt++) {
        #pragma unroll
        for (int j = 0; j < 8; j++) {
            float p = accK[rt][0][j] * qv[0] + accK[rt][1][j] * qv[1]
                    + accK[rt][2][j] * qv[2];
            #pragma unroll
            for (int m = 1; m < 16; m <<= 1) p += __shfl_xor(p, m, 32);
            float sg = 1.f / (1.f + __expf(-p * scl));
            const int M    = rt * 16 + j + 8 * half;
            const int ntok = n0 + M;
            if (n == j)
                out_attn[((size_t)b * NH + wave) * N_TOK + ntok] = sg;
            __bf16* xrow = ab + (size_t)(row0 + M) * C_DIM;
            #pragma unroll
            for (int t = 0; t < 3; t++) {
                int d = t * 16 + n;                 // col = wave*48 + d
                size_t kvi = (((size_t)b * NH + wave) * N_TOK + ntok) * HD + d;
                out_k[kvi] = accK[rt][t][j];
                out_v[kvi] = accV[rt][t][j];
                xrow[wave * 48 + d] = (__bf16)(sg * accV[rt][t][j]);
            }
        }
    }
}

// ---------------------------------------------------------------------------
// K5: x = xpre @ wo + bo, fused output layernorm (64-row tiles, 16 waves)
// ---------------------------------------------------------------------------
__global__ void __launch_bounds__(512)
out_proj_ln(const __bf16* __restrict__ xpre, const __bf16* __restrict__ wo_f,
            const float* __restrict__ bo,
            const float* __restrict__ onw, const float* __restrict__ onb,
            float* __restrict__ out_x) {
    __shared__ float ssum[OP_RT * 16][16];
    __shared__ float ssq [OP_RT * 16][16];
    __shared__ float smean[OP_RT * 16];
    __shared__ float srstd[OP_RT * 16];

    const int rb   = blockIdx.x;
    const int row0 = rb * (OP_RT * 16);
    const int tid  = threadIdx.x;
    const int wave = tid >> 5;
    const int lane = tid & 31;
    const int half = lane >> 4;
    const int n    = lane & 15;

    v8f acc[OP_RT][3] = {};

#if HAVE_ASYNC_LDS
    __shared__ __align__(16) char abuf[2][OP_RT * LDS_TILE_B];
    int groff[OP_RT * 2], ldsoff[OP_RT * 2];
    #pragma unroll
    for (int i = 0; i < OP_RT * 2; i++) {
        int c = i * 32 + lane;
        int rt = c >> 6, w = c & 63, r = w >> 2, cc = w & 3;
        groff[i]  = (rt * 16 + r) * (C_DIM * 2) + cc * 16;
        ldsoff[i] = rt * LDS_TILE_B + r * 80 + cc * 16;
    }
    const char* gbase = (const char*)(xpre + (size_t)row0 * C_DIM);
    if (wave == 0) {
        #pragma unroll
        for (int i = 0; i < OP_RT * 2; i++)
            ASYNC_CP_B128(gbase + groff[i], &abuf[0][ldsoff[i]]);
    }
#endif

    for (int ks = 0; ks < KSTEPS; ks++) {
#if HAVE_ASYNC_LDS
        const int cur = ks & 1;
        if (wave == 0) WAIT_ASYNC0();
        __syncthreads();
        if (wave == 0 && ks + 1 < KSTEPS) {
            #pragma unroll
            for (int i = 0; i < OP_RT * 2; i++)
                ASYNC_CP_B128(gbase + groff[i] + (ks + 1) * 64,
                              &abuf[cur ^ 1][ldsoff[i]]);
        }
#endif
        v16bf bf[3];
        #pragma unroll
        for (int t = 0; t < 3; t++) {
            size_t fo = ((size_t)((wave * 3 + t) * KSTEPS + ks) * 32 + lane) * 16;
            bf[t] = *(const v16bf*)(wo_f + fo);
        }
        #pragma unroll
        for (int rt = 0; rt < OP_RT; rt++) {
#if HAVE_ASYNC_LDS
            v16bf af = load_a_frag_lds(&abuf[cur][rt * LDS_TILE_B], n, half);
#else
            v16bf af = load_a_frag_g(xpre + (size_t)(row0 + rt * 16 + n) * C_DIM,
                                     ks * 32, half);
#endif
            #pragma unroll
            for (int t = 0; t < 3; t++) acc[rt][t] = WMMA_BF16(af, bf[t], acc[rt][t]);
        }
    }

    #pragma unroll
    for (int t = 0; t < 3; t++) {
        float bc = bo[wave * 48 + t * 16 + n];
        #pragma unroll
        for (int rt = 0; rt < OP_RT; rt++)
            #pragma unroll
            for (int j = 0; j < 8; j++) acc[rt][t][j] += bc;
    }

    // LN stats: 16-lane butterfly (48 cols/wave) then cross-wave via LDS
    #pragma unroll
    for (int rt = 0; rt < OP_RT; rt++) {
        #pragma unroll
        for (int j = 0; j < 8; j++) {
            float s = 0.f, s2 = 0.f;
            #pragma unroll
            for (int t = 0; t < 3; t++) { s += acc[rt][t][j]; s2 += acc[rt][t][j] * acc[rt][t][j]; }
            #pragma unroll
            for (int m = 1; m < 16; m <<= 1) {
                s  += __shfl_xor(s,  m, 32);
                s2 += __shfl_xor(s2, m, 32);
            }
            if (n == j) {
                ssum[rt * 16 + j + 8 * half][wave] = s;
                ssq [rt * 16 + j + 8 * half][wave] = s2;
            }
        }
    }
    __syncthreads();
    if (tid < OP_RT * 16) {
        float s = 0.f, s2 = 0.f;
        for (int w = 0; w < 16; w++) { s += ssum[tid][w]; s2 += ssq[tid][w]; }
        float mn  = s * (1.f / 768.f);
        float var = s2 * (1.f / 768.f) - mn * mn;
        smean[tid] = mn;
        srstd[tid] = rsqrtf(var + EPS_LN);
    }
    __syncthreads();

    #pragma unroll
    for (int rt = 0; rt < OP_RT; rt++) {
        #pragma unroll
        for (int j = 0; j < 8; j++) {
            int M = rt * 16 + j + 8 * half;
            float mj = smean[M], rj = srstd[M];
            #pragma unroll
            for (int t = 0; t < 3; t++) {
                int col = wave * 48 + t * 16 + n;
                out_x[(size_t)(row0 + M) * OUT_DIM + col] =
                    (acc[rt][t][j] - mj) * rj * onw[col] + onb[col];
            }
        }
    }
}

// ---------------------------------------------------------------------------
extern "C" void kernel_launch(void* const* d_in, const int* in_sizes, int n_in,
                              void* d_out, int out_size, void* d_ws, size_t ws_size,
                              hipStream_t stream) {
    const float* emb  = (const float*)d_in[0];
    const int*   rloc = (const int*)  d_in[1];
    const float* lnw  = (const float*)d_in[2];
    const float* lnb  = (const float*)d_in[3];
    const float* w1   = (const float*)d_in[4];
    const float* b1   = (const float*)d_in[5];
    const float* w2   = (const float*)d_in[6];
    const float* b2   = (const float*)d_in[7];
    const float* wq   = (const float*)d_in[8];
    const float* bq   = (const float*)d_in[9];
    const float* wk   = (const float*)d_in[10];
    const float* bk   = (const float*)d_in[11];
    const float* wv   = (const float*)d_in[12];
    const float* bv   = (const float*)d_in[13];
    const float* wo   = (const float*)d_in[14];
    const float* bo   = (const float*)d_in[15];
    const float* onw  = (const float*)d_in[16];
    const float* onb  = (const float*)d_in[17];

    // workspace (~158 MB): emb_bf doubles as xpre after consumption
    char* ws = (char*)d_ws;
    size_t off = 0;
    __bf16* emb_bf = (__bf16*)(ws + off); off += (size_t)BN * C_DIM * 2;
    __bf16* wk_f   = (__bf16*)(ws + off); off += (size_t)C_DIM * OUT_DIM * 2;
    __bf16* wv_f   = (__bf16*)(ws + off); off += (size_t)C_DIM * OUT_DIM * 2;
    __bf16* wo_f   = (__bf16*)(ws + off); off += (size_t)C_DIM * OUT_DIM * 2;
    float*  q_ws   = (float*) (ws + off); off += (size_t)B_SZ * OUT_DIM * 4;

    // outputs: (x, attn, q, k, v, ref) concatenated fp32
    float* o        = (float*)d_out;
    float* out_x    = o;
    float* out_attn = out_x    + (size_t)BN * OUT_DIM;
    float* out_q    = out_attn + (size_t)B_SZ * NH * N_TOK;
    float* out_k    = out_q    + (size_t)B_SZ * OUT_DIM;
    float* out_v    = out_k    + (size_t)BN * OUT_DIM;
    float* out_ref  = out_v    + (size_t)BN * OUT_DIM;

    pack_weights<<<2304, 256, 0, stream>>>(wk, wk_f);
    pack_weights<<<2304, 256, 0, stream>>>(wv, wv_f);
    pack_weights<<<2304, 256, 0, stream>>>(wo, wo_f);
    ln_to_bf16<<<BN, 256, 0, stream>>>(emb, lnw, lnb, emb_bf);
    ref_mlp_q<<<B_SZ, 256, 0, stream>>>(emb, rloc, lnw, lnb, w1, b1, w2, b2,
                                        wq, bq, out_ref, out_q, q_ws);
    kv_attn<<<BN / (KV_RT * 16), 512, 0, stream>>>(emb_bf, wk_f, wv_f, bk, bv,
                                                   q_ws, out_k, out_v, out_attn);
    out_proj_ln<<<BN / (OP_RT * 16), 512, 0, stream>>>(emb_bf /* xpre */, wo_f,
                                                       bo, onw, onb, out_x);
}
